// Mamba3Mixer_82471962018258
// MI455X (gfx1250) — matrix operand order repaired and checked by reading, not compiled, not run on hardware
//
#include <hip/hip_runtime.h>

// ---------------------------------------------------------------------------
// Mamba3 mixer for MI455X (gfx1250): bf16 WMMA GEMMs + LDS-resident SSM scan.
// ---------------------------------------------------------------------------

typedef __attribute__((ext_vector_type(16))) __bf16 v16bf;
typedef __attribute__((ext_vector_type(8)))  float  v8f;
typedef __attribute__((ext_vector_type(4)))  __bf16 bf16x4;   // 8 B
typedef __attribute__((ext_vector_type(8)))  __bf16 bf16x8;   // 16 B

union FragAB { v16bf v; unsigned u[8]; };

#define B_SZ      2
#define SEQLEN    512
#define D_MODEL   1024
#define D_INNER   2048
#define N_HEADS   32
#define HEAD_DIM  64
#define D_STATE   64
#define PROJ_DIM  8224          // 2*D_INNER + 2*N_HEADS*D_STATE + N_HEADS
#define ROWS      (B_SZ * SEQLEN)   // 1024

#define TILE_M 128
#define TILE_N 64
#define TILE_K 32
#define LDS_STRIDE 40           // bf16 elems per row: 80 B, 16B-aligned rows

// ---------------------------------------------------------------------------
// Generic fp32-in / fp32-out GEMM, bf16 WMMA core with fp32 accumulation.
// C[M,N] = A[M,K] @ B[K,N].  M % 128 == 0, K % 32 == 0, N guarded on store.
// B column accesses are clamped (not zeroed): C[:,n] depends only on B[:,n],
// and out-of-range output columns are never stored, so clamped reads are safe
// and keep the staging loads branch-free / clause-groupable.
// ---------------------------------------------------------------------------
__global__ __launch_bounds__(256)
void gemm_bf16_wmma(const float* __restrict__ A, const float* __restrict__ B,
                    float* __restrict__ C, int M, int N, int K)
{
    __shared__ __bf16 Alds[TILE_M][LDS_STRIDE];   // [m][k]
    __shared__ __bf16 Blds[TILE_N][LDS_STRIDE];   // [n][k] (K-transposed)

    const int tid  = threadIdx.x;
    const int lane = tid & 31;
    const int wave = tid >> 5;     // 8 waves
    const int wm   = wave >> 1;    // 0..3  -> 32-row strips
    const int wn   = wave & 1;     // 0..1  -> 32-col strips
    const int hi   = lane >> 4;    // lane half
    const int l15  = lane & 15;

    const int mBase = blockIdx.y * TILE_M;
    const int nBase = blockIdx.x * TILE_N;

    // B staging coords: thread -> (n column, 8-deep k segment), clamped col
    const int bnc = tid & 63;
    const int bkq = tid >> 6;           // 0..3 -> k = bkq*8 .. +7
    int bgn = nBase + bnc;
    if (bgn > N - 1) bgn = N - 1;       // clamp: safe, see header comment

    v8f acc[2][2];
    const v8f vzero = {0.f,0.f,0.f,0.f,0.f,0.f,0.f,0.f};
    acc[0][0] = vzero; acc[0][1] = vzero; acc[1][0] = vzero; acc[1][1] = vzero;

    float4 fa[4];
    float  fb[8];

    // ---- preload slab k0 = 0 into registers ----
    #pragma unroll
    for (int r = 0; r < 4; ++r) {
        int e4 = r * 256 + tid, row = e4 >> 3, c4 = e4 & 7;
        fa[r] = *(const float4*)(A + (size_t)(mBase + row) * K + c4 * 4);
    }
    {
        const float* bp = B + (size_t)(bkq * 8) * N + bgn;
        #pragma unroll
        for (int j = 0; j < 8; ++j) fb[j] = bp[(size_t)j * N];
    }

    for (int k0 = 0; k0 < K; k0 += TILE_K) {
        // ---- commit staged registers to LDS (packed bf16 stores) ----
        #pragma unroll
        for (int r = 0; r < 4; ++r) {
            int e4 = r * 256 + tid, row = e4 >> 3, c4 = e4 & 7;
            bf16x4 v;
            v[0] = (__bf16)fa[r].x; v[1] = (__bf16)fa[r].y;
            v[2] = (__bf16)fa[r].z; v[3] = (__bf16)fa[r].w;
            *(bf16x4*)&Alds[row][c4 * 4] = v;        // 8B-aligned
        }
        {
            bf16x8 v;
            #pragma unroll
            for (int j = 0; j < 8; ++j) v[j] = (__bf16)fb[j];
            *(bf16x8*)&Blds[bnc][bkq * 8] = v;       // 16B-aligned
        }
        __syncthreads();

        // ---- prefetch next slab into registers (overlaps WMMAs) ----
        const int kn = k0 + TILE_K;
        if (kn < K) {
            #pragma unroll
            for (int r = 0; r < 4; ++r) {
                int e4 = r * 256 + tid, row = e4 >> 3, c4 = e4 & 7;
                fa[r] = *(const float4*)(A + (size_t)(mBase + row) * K + kn + c4 * 4);
            }
            const float* bp = B + (size_t)(kn + bkq * 8) * N + bgn;
            #pragma unroll
            for (int j = 0; j < 8; ++j) fb[j] = bp[(size_t)j * N];
        }

        // ---- build fragments: A and B both in contiguous order k = 16*hi + s ----
        FragAB af[2], bf2[2];
        #pragma unroll
        for (int t = 0; t < 2; ++t) {
            int mrow = wm * 32 + t * 16 + l15;
            int ncol = wn * 32 + t * 16 + l15;
            #pragma unroll
            for (int v = 0; v < 8; ++v) {
                int ka = hi * 16 + 2 * v;
                int kb = hi * 16 + 2 * v;
                af[t].u[v]  = *(const unsigned*)&Alds[mrow][ka];
                bf2[t].u[v] = *(const unsigned*)&Blds[ncol][kb];
            }
        }
        #pragma unroll
        for (int i = 0; i < 2; ++i)
            #pragma unroll
            for (int j = 0; j < 2; ++j)
                acc[i][j] = __builtin_amdgcn_wmma_f32_16x16x32_bf16(
                    false, af[i].v, false, bf2[j].v, (short)0, acc[i][j],
                    false, false);
        __syncthreads();
    }

    // ---- store: C/D layout -> VGPR r holds (m = r + 8*hi, n = lane&15) ----
    #pragma unroll
    for (int i = 0; i < 2; ++i) {
        #pragma unroll
        for (int j = 0; j < 2; ++j) {
            int gcol = nBase + wn * 32 + j * 16 + l15;
            if (gcol < N) {
                #pragma unroll
                for (int r = 0; r < 8; ++r) {
                    int grow = mBase + wm * 32 + i * 16 + r + 8 * hi;
                    C[(size_t)grow * N + gcol] = acc[i][j][r];
                }
            }
        }
    }
}

// ---------------------------------------------------------------------------
// SSM scan: one block per (batch, head). 256 threads; each owns 16 state
// elements h[n0..n0+15][p]. Double-buffered LDS gives Bprev/xprev for free.
// ---------------------------------------------------------------------------
__global__ __launch_bounds__(256)
void mamba_scan(const float* __restrict__ proj, const float* __restrict__ dt_bias,
                const float* __restrict__ A_log, float* __restrict__ yraw)
{
    const int bh  = blockIdx.x;          // 0..63
    const int b   = bh >> 5;
    const int h   = bh & 31;
    const int tid = threadIdx.x;

    __shared__ float Bs[2][D_STATE];
    __shared__ float Cs[2][D_STATE];
    __shared__ float Xs[2][HEAD_DIM];
    __shared__ float red[256];

    if (tid < 128) {                      // zero both buffers (t=0 padding)
        int bi = tid >> 6, e = tid & 63;
        Bs[bi][e] = 0.f; Cs[bi][e] = 0.f; Xs[bi][e] = 0.f;
    }
    __syncthreads();

    const float bias = dt_bias[h];
    const float negA = -__expf(A_log[h]);

    // rope roles: tid<32 -> B pair, 32..63 -> C pair
    const int  pairi = tid & 31;
    const float invf = __expf(-(float)pairi * (0.03125f * __logf(10000.0f)));

    const int p  = tid & 63;
    const int n0 = (tid >> 6) * 16;

    float hst[16];
    #pragma unroll
    for (int j = 0; j < 16; ++j) hst[j] = 0.f;

    float tsum = 0.f;
    const float* prow = proj + (size_t)b * SEQLEN * PROJ_DIM;

    for (int t = 0; t < SEQLEN; ++t) {
        const int cur = t & 1, prv = cur ^ 1;
        const float* rp = prow + (size_t)t * PROJ_DIM;

        // dt / alpha / beta / gamma, inclusive cumsum (uniform scalar work)
        float xv    = rp[2 * D_INNER + 2 * N_HEADS * D_STATE + h] + bias;
        float dt    = (xv > 20.f) ? xv : log1pf(__expf(xv));
        tsum       += dt;
        float alpha = __expf(dt * negA);
        float beta  = 0.5f * dt * alpha;
        float gamma = 0.5f * dt;

        if (tid < 32) {                         // RoPE(B_t)
            float th = tsum * invf;
            float c = __cosf(th), s = __sinf(th);
            const float* bp = rp + 2 * D_INNER + h * D_STATE;
            float b0 = bp[2 * pairi], b1 = bp[2 * pairi + 1];
            Bs[cur][2 * pairi]     = b0 * c - b1 * s;
            Bs[cur][2 * pairi + 1] = b0 * s + b1 * c;
            __builtin_prefetch(bp + PROJ_DIM + 2 * pairi, 0, 0);
        } else if (tid < 64) {                  // RoPE(C_t)
            float th = tsum * invf;
            float c = __cosf(th), s = __sinf(th);
            const float* cp = rp + 2 * D_INNER + N_HEADS * D_STATE + h * D_STATE;
            float c0 = cp[2 * pairi], c1 = cp[2 * pairi + 1];
            Cs[cur][2 * pairi]     = c0 * c - c1 * s;
            Cs[cur][2 * pairi + 1] = c0 * s + c1 * c;
            __builtin_prefetch(cp + PROJ_DIM + 2 * pairi, 0, 0);
        } else if (tid < 128) {                 // x_t (raw)
            int pp = tid - 64;
            const float* xp = rp + D_INNER + h * HEAD_DIM;
            Xs[cur][pp] = xp[pp];
            __builtin_prefetch(xp + PROJ_DIM + pp, 0, 0);
        }
        __syncthreads();

        // state update + partial y = C . h
        float xc = Xs[cur][p], xp_ = Xs[prv][p];
        float yacc = 0.f;
        #pragma unroll
        for (int j = 0; j < 16; ++j) {
            int n = n0 + j;
            float hv = alpha * hst[j] + beta * (Bs[prv][n] * xp_)
                                      + gamma * (Bs[cur][n] * xc);
            hst[j] = hv;
            yacc  += Cs[cur][n] * hv;
        }
        red[tid] = yacc;
        __syncthreads();

        if (tid < 64) {
            float y = red[tid] + red[64 + tid] + red[128 + tid] + red[192 + tid];
            yraw[(size_t)(b * SEQLEN + t) * D_INNER + h * HEAD_DIM + tid] = y;
        }
        // next iter's first barrier orders red[] reuse + buffer swap
    }
}

// ---------------------------------------------------------------------------
// y = yraw * silu(z); RMS-norm over D_INNER; scale by norm_weight.
// One block per (b,l) row.
// ---------------------------------------------------------------------------
__global__ __launch_bounds__(256)
void gate_rmsnorm(const float* __restrict__ proj, const float* __restrict__ yraw,
                  const float* __restrict__ nw, float* __restrict__ yn)
{
    const int row = blockIdx.x;
    const int tid = threadIdx.x;
    const float* zr = proj + (size_t)row * PROJ_DIM;     // z = proj[..., :D_INNER]
    const float* yr = yraw + (size_t)row * D_INNER;

    float v[8];
    float ss = 0.f;
    #pragma unroll
    for (int i = 0; i < 8; ++i) {
        int idx = tid + i * 256;
        float z   = zr[idx];
        float sig = 1.f / (1.f + __expf(-z));
        float val = yr[idx] * z * sig;
        v[i] = val;
        ss  += val * val;
    }
    __shared__ float red[256];
    red[tid] = ss;
    __syncthreads();
    #pragma unroll
    for (int s = 128; s > 0; s >>= 1) {
        if (tid < s) red[tid] += red[tid + s];
        __syncthreads();
    }
    float rinv = rsqrtf(red[0] * (1.0f / (float)D_INNER) + 1e-6f);

    float* yo = yn + (size_t)row * D_INNER;
    #pragma unroll
    for (int i = 0; i < 8; ++i) {
        int idx = tid + i * 256;
        yo[idx] = v[i] * rinv * nw[idx];
    }
}

// ---------------------------------------------------------------------------
extern "C" void kernel_launch(void* const* d_in, const int* in_sizes, int n_in,
                              void* d_out, int out_size, void* d_ws, size_t ws_size,
                              hipStream_t stream)
{
    (void)in_sizes; (void)n_in; (void)out_size; (void)ws_size;

    const float* hidden = (const float*)d_in[0];   // (2,512,1024)
    const float* W_in   = (const float*)d_in[1];   // (1024,8224)
    const float* dtb    = (const float*)d_in[2];   // (32,)
    const float* A_log  = (const float*)d_in[3];   // (32,)
    const float* nw     = (const float*)d_in[4];   // (2048,)
    const float* W_out  = (const float*)d_in[5];   // (2048,1024)
    float*       out    = (float*)d_out;           // (2,512,1024) fp32

    // workspace: proj (1024x8224) | yraw (1024x2048) | ynorm (1024x2048)
    float* proj  = (float*)d_ws;
    float* yraw  = proj + (size_t)ROWS * PROJ_DIM;
    float* ynorm = yraw + (size_t)ROWS * D_INNER;

    dim3 blk(256);

    // 1) proj = hidden @ W_in : M=1024, N=8224, K=1024
    gemm_bf16_wmma<<<dim3((PROJ_DIM + TILE_N - 1) / TILE_N, ROWS / TILE_M),
                     blk, 0, stream>>>(hidden, W_in, proj, ROWS, PROJ_DIM, D_MODEL);

    // 2) sequential SSM scan -> yraw
    mamba_scan<<<dim3(B_SZ * N_HEADS), blk, 0, stream>>>(proj, dtb, A_log, yraw);

    // 3) gate + RMS norm -> ynorm
    gate_rmsnorm<<<dim3(ROWS), blk, 0, stream>>>(proj, yraw, nw, ynorm);

    // 4) out = ynorm @ W_out : M=1024, N=1024, K=2048
    gemm_bf16_wmma<<<dim3(D_MODEL / TILE_N, ROWS / TILE_M),
                     blk, 0, stream>>>(ynorm, W_out, out, ROWS, D_MODEL, D_INNER);
}
